// SpatialTransformer_89541478187569
// MI455X (gfx1250) — compile-verified, matrix operand order checked by reading, not run on hardware
//
#include <hip/hip_runtime.h>
#include <cstdint>
#include <cstddef>

// ---------------------------------------------------------------------------
// CDNA5 (gfx1250) SpatialTransformer forward.
// All heavy GEMMs run through a wave32 WMMA kernel (v_wmma_f32_16x16x32_f16)
// with vectorized b128 staging and register-level software pipelining.
// ---------------------------------------------------------------------------

typedef __attribute__((ext_vector_type(16))) _Float16 v16h;
typedef __attribute__((ext_vector_type(8)))  float    v8f;

#define B_    4
#define C_    640
#define HW_   1024
#define NT_   4096      // B_*HW_
#define D_    640
#define HEADS_ 8
#define DH_   80
#define CTX_  77
#define CTXD_ 768
#define FF1_  5120
#define FF2_  2560

// ------------------------------ WMMA GEMM ----------------------------------
// D = A(MxK) * B(KxN) [+bias[n]] [+res]  with two-level (b,h) batching.
// A,B,res are f32; staged to f16 in LDS; accumulate f32 via v_wmma.
struct GemmP {
  const float* A; const float* Bm; const float* bias; const float* res;
  float* D;
  int M, N, K;
  int lda, ldb, ldd, ldres;
  long long sAb, sAh, sBb, sBh, sDb, sDh, sResb;
  int heads;      // z -> b = z/heads, h = z%heads
  int bIsNK;      // B stored row-major as N x K (used for q @ k^T)
  int storeTrans; // store D[n*ldd + m] (proj_out)
};

#define LDS_LD 36  // 32 + 4 halves pad (rows stay 8B aligned: 36*2 = 72)

__device__ __forceinline__ v16h load_frag16x32(const _Float16* tile, int row0, int lane) {
  // 16-bit A/B fragment layout (ISA 7.12.2):
  // lanes 0-15: m=lane,   vgpr0-3 -> K 0..7,  vgpr4-7 -> K 16..23
  // lanes16-31: m=lane-16,vgpr0-3 -> K 8..15, vgpr4-7 -> K 24..31
  const _Float16* rp = tile + (size_t)(row0 + (lane & 15)) * LDS_LD;
  const int kb = (lane >> 4) << 3;
  union { v16h v; unsigned u[8]; } f;
#pragma unroll
  for (int i = 0; i < 8; ++i) {
    int k = ((i >> 2) << 4) + kb + ((i & 3) << 1);
    f.u[i] = *reinterpret_cast<const unsigned*>(rp + k);
  }
  return f.v;
}

// Row-major staging: thread owns 16 contiguous inner-dim elems of one tile row.
// tid>>1 = row (0..63), (tid&1)*16 = column base.
__device__ __forceinline__ void load_rowmajor16(const float* __restrict__ src0,
                                                int nrowsValid, int innerValid,
                                                int ld, int row0, int col0,
                                                int tid, float (&r)[16]) {
  const int row = tid >> 1, cb = (tid & 1) << 4;
  const int gr = row0 + row, gc = col0 + cb;
  const float* src = src0 + (size_t)gr * ld + gc;
  if (gr < nrowsValid && gc + 16 <= innerValid) {
#pragma unroll
    for (int j = 0; j < 4; ++j) {
      float4 v = *reinterpret_cast<const float4*>(src + 4 * j);
      r[4*j] = v.x; r[4*j+1] = v.y; r[4*j+2] = v.z; r[4*j+3] = v.w;
    }
  } else {
#pragma unroll
    for (int i = 0; i < 16; ++i)
      r[i] = (gr < nrowsValid && (gc + i) < innerValid) ? src[i] : 0.f;
  }
}

__device__ __forceinline__ void store_rowmajor16(_Float16 (&tile)[64][LDS_LD],
                                                 int tid, const float (&r)[16]) {
  const int row = tid >> 1, cb = (tid & 1) << 4;
  _Float16* dst = &tile[row][cb];
#pragma unroll
  for (int j = 0; j < 4; ++j) {
    union { _Float16 h[4]; unsigned long long u; } t;
#pragma unroll
    for (int e = 0; e < 4; ++e) t.h[e] = (_Float16)r[4*j+e];
    *reinterpret_cast<unsigned long long*>(dst + 4*j) = t.u;  // ds_store_b64
  }
}

// K-major B (weights, K x N): thread owns 16 contiguous n for one k.
// tid>>2 = k (0..31), (tid&3)*16 = n base. Written transposed into Bs[n][k].
__device__ __forceinline__ void load_kn16(const float* __restrict__ Bm,
                                          int N, int K, int ldb,
                                          int n0, int k0, int tid, float (&r)[16]) {
  const int k = tid >> 2, nb = (tid & 3) << 4;
  const int gk = k0 + k, gn = n0 + nb;
  const float* src = Bm + (size_t)gk * ldb + gn;
  if (gk < K && gn + 16 <= N) {
#pragma unroll
    for (int j = 0; j < 4; ++j) {
      float4 v = *reinterpret_cast<const float4*>(src + 4 * j);
      r[4*j] = v.x; r[4*j+1] = v.y; r[4*j+2] = v.z; r[4*j+3] = v.w;
    }
  } else {
#pragma unroll
    for (int i = 0; i < 16; ++i)
      r[i] = (gk < K && (gn + i) < N) ? src[i] : 0.f;
  }
}

__device__ __forceinline__ void store_kn16(_Float16 (&Bs)[64][LDS_LD],
                                           int tid, const float (&r)[16]) {
  const int k = tid >> 2, nb = (tid & 3) << 4;
#pragma unroll
  for (int i = 0; i < 16; ++i) Bs[nb + i][k] = (_Float16)r[i];
}

__global__ __launch_bounds__(128) void gemm_f16wmma(GemmP p) {
  __shared__ _Float16 As[64][LDS_LD];
  __shared__ _Float16 Bs[64][LDS_LD];

  const int z = blockIdx.z;
  const int b = z / p.heads, hh = z % p.heads;
  const float* A  = p.A  + (long long)b * p.sAb + (long long)hh * p.sAh;
  const float* Bm = p.Bm + (long long)b * p.sBb + (long long)hh * p.sBh;
  float*       Dp = p.D  + (long long)b * p.sDb + (long long)hh * p.sDh;
  const float* Rp = p.res ? p.res + (long long)b * p.sResb : nullptr;

  const int m0 = blockIdx.y * 64;
  const int n0 = blockIdx.x * 64;
  const int tid = threadIdx.x;
  const int wave = tid >> 5, lane = tid & 31;
  const int wm = (wave >> 1) * 32;   // wave grid 2x2, 32x32 per wave
  const int wn = (wave & 1) * 32;

  const v8f vzero = {0.f,0.f,0.f,0.f,0.f,0.f,0.f,0.f};
  v8f acc[2][2] = {{vzero, vzero}, {vzero, vzero}};

  float ra[16], rb[16];
  // prologue: prefetch tile 0 into registers
  load_rowmajor16(A, p.M, p.K, p.lda, m0, 0, tid, ra);
  if (p.bIsNK) load_rowmajor16(Bm, p.N, p.K, p.ldb, n0, 0, tid, rb);
  else         load_kn16(Bm, p.N, p.K, p.ldb, n0, 0, tid, rb);

  for (int k0 = 0; k0 < p.K; k0 += 32) {
    // commit current tile to LDS
    store_rowmajor16(As, tid, ra);
    if (p.bIsNK) store_rowmajor16(Bs, tid, rb);
    else         store_kn16(Bs, tid, rb);
    __syncthreads();

    // issue next tile's global loads (overlap with ds+wmma below)
    const int kn = k0 + 32;
    if (kn < p.K) {
      load_rowmajor16(A, p.M, p.K, p.lda, m0, kn, tid, ra);
      if (p.bIsNK) load_rowmajor16(Bm, p.N, p.K, p.ldb, n0, kn, tid, rb);
      else         load_kn16(Bm, p.N, p.K, p.ldb, n0, kn, tid, rb);
    }

    v16h a0 = load_frag16x32(&As[0][0], wm,      lane);
    v16h a1 = load_frag16x32(&As[0][0], wm + 16, lane);
    v16h b0 = load_frag16x32(&Bs[0][0], wn,      lane);
    v16h b1 = load_frag16x32(&Bs[0][0], wn + 16, lane);
    acc[0][0] = __builtin_amdgcn_wmma_f32_16x16x32_f16(false, a0, false, b0, (short)0, acc[0][0], false, false);
    acc[0][1] = __builtin_amdgcn_wmma_f32_16x16x32_f16(false, a0, false, b1, (short)0, acc[0][1], false, false);
    acc[1][0] = __builtin_amdgcn_wmma_f32_16x16x32_f16(false, a1, false, b0, (short)0, acc[1][0], false, false);
    acc[1][1] = __builtin_amdgcn_wmma_f32_16x16x32_f16(false, a1, false, b1, (short)0, acc[1][1], false, false);
    __syncthreads();
  }

  // C/D layout: lane 0-15 -> n=lane, vgpr r -> m=r ; lane 16-31 -> m=8+r
  const int nCol  = lane & 15;
  const int rowOf = (lane >> 4) * 8;
#pragma unroll
  for (int ti = 0; ti < 2; ++ti)
#pragma unroll
    for (int tj = 0; tj < 2; ++tj) {
      int mb = m0 + wm + ti * 16 + rowOf;
      int n  = n0 + wn + tj * 16 + nCol;
#pragma unroll
      for (int r = 0; r < 8; ++r) {
        int m = mb + r;
        if (m < p.M && n < p.N) {
          float v = acc[ti][tj][r];
          if (p.bias) v += p.bias[n];
          long long idx = p.storeTrans ? ((long long)n * p.ldd + m)
                                       : ((long long)m * p.ldd + n);
          if (Rp) {
            long long ridx = p.storeTrans ? ((long long)n * p.ldres + m)
                                          : ((long long)m * p.ldres + n);
            v += Rp[ridx];
          }
          Dp[idx] = v;
        }
      }
    }
}

// --------------------------- reduction helpers -----------------------------
__device__ __forceinline__ float blkSum(float v, float* red) {
  int t = threadIdx.x; red[t] = v; __syncthreads();
  for (int s = 128; s > 0; s >>= 1) { if (t < s) red[t] += red[t + s]; __syncthreads(); }
  float r = red[0]; __syncthreads(); return r;
}
__device__ __forceinline__ float blkMax(float v, float* red) {
  int t = threadIdx.x; red[t] = v; __syncthreads();
  for (int s = 128; s > 0; s >>= 1) { if (t < s) red[t] = fmaxf(red[t], red[t + s]); __syncthreads(); }
  float r = red[0]; __syncthreads(); return r;
}
__device__ __forceinline__ float blkMin(float v, float* red) {
  int t = threadIdx.x; red[t] = v; __syncthreads();
  for (int s = 128; s > 0; s >>= 1) { if (t < s) red[t] = fminf(red[t], red[t + s]); __syncthreads(); }
  float r = red[0]; __syncthreads(); return r;
}

// ------------------------- GroupNorm + transpose ---------------------------
// x:[B,C,H*W] -> out_t:[B,HW,C], 32 groups of 20 channels, eps 1e-6
__global__ __launch_bounds__(256) void gn_transpose(const float* __restrict__ x,
                                                    const float* __restrict__ g,
                                                    const float* __restrict__ be,
                                                    float* __restrict__ out) {
  __shared__ float red[256];
  int blk = blockIdx.x;
  int b = blk >> 5, grp = blk & 31;
  int c0 = grp * 20;
  const float* xb = x + ((size_t)b * C_ + c0) * HW_;
  int tid = threadIdx.x;
  float su = 0.f, sq = 0.f;
  const float4* xb4 = reinterpret_cast<const float4*>(xb);
  for (int i = tid; i < 20 * HW_ / 4; i += 256) {
    float4 v = xb4[i];
    su += v.x + v.y + v.z + v.w;
    sq += v.x * v.x + v.y * v.y + v.z * v.z + v.w * v.w;
  }
  float mean = blkSum(su, red) * (1.f / 20480.f);
  float var  = blkSum(sq, red) * (1.f / 20480.f) - mean * mean;
  float rstd = rsqrtf(var + 1e-6f);
  for (int i = tid; i < 20 * HW_; i += 256) {
    int cc = i >> 10, pp = i & 1023, c = c0 + cc;
    out[((size_t)b * HW_ + pp) * C_ + c] = (xb[i] - mean) * rstd * g[c] + be[c];
  }
}

// ------------------------------- LayerNorm ---------------------------------
__global__ __launch_bounds__(256) void ln_rows(const float* __restrict__ in,
                                               const float* __restrict__ g,
                                               const float* __restrict__ be,
                                               float* __restrict__ out, int D) {
  __shared__ float red[256];
  const float* r = in + (size_t)blockIdx.x * D;
  float* o = out + (size_t)blockIdx.x * D;
  int tid = threadIdx.x;
  float su = 0.f, sq = 0.f;
  for (int j = tid; j < D; j += 256) { float v = r[j]; su += v; sq += v * v; }
  float mean = blkSum(su, red) / (float)D;
  float var  = blkSum(sq, red) / (float)D - mean * mean;
  float rstd = rsqrtf(var + 1e-5f);
  for (int j = tid; j < D; j += 256) o[j] = (r[j] - mean) * rstd * g[j] + be[j];
}

// ------------------------- scaled masked softmax ---------------------------
__global__ __launch_bounds__(256) void softmax_rows(float* __restrict__ buf, int m, int ld,
                                                    float scale, const int* __restrict__ mask,
                                                    int rowsPerB, int maskLen) {
  __shared__ float red[256];
  size_t row = blockIdx.x;
  float* r = buf + row * (size_t)ld;
  const int* mrow = mask ? (mask + (row / (size_t)rowsPerB) * (size_t)maskLen) : nullptr;
  int tid = threadIdx.x;
  const float NEG = -3.402823466e38f;
  float lv[4];   // m <= 1024, 256 threads -> at most 4 elements per thread
  int cnt = 0;
  float mx = NEG;
  for (int j = tid; j < m; j += 256) {
    float v = r[j] * scale;
    if (mrow && mrow[j] == 0) v = NEG;
    lv[cnt++] = v;
    mx = fmaxf(mx, v);
  }
  mx = blkMax(mx, red);
  float sum = 0.f;
#pragma unroll
  for (int i = 0; i < 4; ++i)
    if (i < cnt) { lv[i] = expf(lv[i] - mx); sum += lv[i]; }
  sum = blkSum(sum, red);
  float inv = 1.f / sum;
#pragma unroll
  for (int i = 0; i < 4; ++i)
    if (i < cnt) r[tid + 256 * i] = lv[i] * inv;
}

// --------------------------------- GEGLU -----------------------------------
__global__ __launch_bounds__(256) void geglu_k(const float* __restrict__ y,
                                               float* __restrict__ o, long long total) {
  long long idx = (long long)blockIdx.x * 256 + threadIdx.x;
  if (idx >= total) return;
  long long rr = idx / FF2_, c = idx % FF2_;
  float a = y[rr * FF1_ + c];
  float g = y[rr * FF1_ + FF2_ + c];
  float ge = 0.5f * g * (1.f + erff(g * 0.70710678118654752f));
  o[idx] = a * ge;
}

// -------------------- attention mean over heads (ctx) ----------------------
// probs:[B,H,HW,80] (valid j<77) -> am:[B,HW,77]
__global__ __launch_bounds__(256) void attn_mean_k(const float* __restrict__ probs,
                                                   float* __restrict__ am) {
  long long idx = (long long)blockIdx.x * 256 + threadIdx.x;
  if (idx >= (long long)B_ * HW_ * CTX_) return;
  int j = (int)(idx % CTX_);
  long long t = idx / CTX_;
  int i = (int)(t % HW_);
  int b = (int)(t / HW_);
  float s = 0.f;
#pragma unroll
  for (int h = 0; h < HEADS_; ++h)
    s += probs[(((long long)(b * HEADS_ + h) * HW_) + i) * DH_ + j];
  am[idx] = s * (1.f / HEADS_);
}

// ------------------- gather ph/eot attention columns -----------------------
__global__ __launch_bounds__(256) void gather_ph_k(const float* __restrict__ am,
                                                   const int* __restrict__ phIdx,
                                                   const int* __restrict__ eotIdx,
                                                   float* __restrict__ ph,
                                                   float* __restrict__ eot) {
  int idx = blockIdx.x * 256 + threadIdx.x;
  if (idx >= NT_) return;
  int b = idx >> 10;
  ph[idx]  = am[(size_t)idx * CTX_ + phIdx[b]];
  eot[idx] = am[(size_t)idx * CTX_ + eotIdx[b]];
}

// --------------------------- regularization loss ---------------------------
__global__ __launch_bounds__(256) void loss_k(const float* __restrict__ ph,
                                              const float* __restrict__ eot,
                                              float* __restrict__ out) {
  __shared__ float red[256];
  __shared__ float accS;
  int tid = threadIdx.x;
  if (tid == 0) accS = 0.f;
  __syncthreads();
  for (int b = 0; b < B_; ++b) {
    const float* p = ph + b * HW_;
    const float* e = eot + b * HW_;
    float mp = -3.402823466e38f, me = -3.402823466e38f;
    for (int j = tid; j < HW_; j += 256) { mp = fmaxf(mp, p[j]); me = fmaxf(me, e[j]); }
    mp = blkMax(mp, red);
    me = blkMax(me, red);
    float part = 0.f;
    for (int j = tid; j < HW_; j += 256) {
      float d = p[j] / mp - e[j] / me;
      part += d * d;
    }
    part = blkSum(part, red);
    if (tid == 0) accS += part;
    __syncthreads();
  }
  if (tid == 0) out[0] = accS / (float)NT_;
}

// ------------------------------ Otsu mask ----------------------------------
__global__ __launch_bounds__(256) void otsu_mask_k(const float* __restrict__ ph,
                                                   int* __restrict__ mask) {
  __shared__ float red[256];
  int b = blockIdx.x, tid = threadIdx.x;
  const float* m = ph + b * HW_;
  float lo = 3.402823466e38f, hi = -3.402823466e38f;
  for (int j = tid; j < HW_; j += 256) { lo = fminf(lo, m[j]); hi = fmaxf(hi, m[j]); }
  lo = blkMin(lo, red);
  hi = blkMax(hi, red);
  float inv = 1.f / (hi - lo);
  float tot = 0.f;
  for (int j = tid; j < HW_; j += 256) tot += (m[j] - lo) * inv;
  tot = blkSum(tot, red);
  float thr = 0.f, maxg = 0.f;
  for (int t = 0; t < 10; ++t) {
    float tv = 0.1f * (float)t;
    float cnt = 0.f, lsum = 0.f;
    for (int j = tid; j < HW_; j += 256) {
      float mn = (m[j] - lo) * inv;
      if (mn < tv) { cnt += 1.f; lsum += mn; }
    }
    cnt  = blkSum(cnt, red);
    lsum = blkSum(lsum, red);
    float lowNum  = cnt * (1.f / HW_);
    float highNum = (HW_ - cnt) * (1.f / HW_);
    float lm = lsum / cnt;                       // NaN when cnt==0 (matches ref)
    float hm = (tot - lsum) / ((float)HW_ - cnt);
    float g = lowNum * highNum * (lm - hm) * (lm - hm);
    if (g > maxg) { maxg = g; thr = tv; }        // NaN compares false
  }
  for (int j = tid; j < HW_; j += 256) {
    float mn = (m[j] - lo) * inv;
    float val = (mn < thr) ? 0.f : ((mn > thr) ? 1.f : mn);
    mask[b * HW_ + j] = (val != 0.f) ? 1 : 0;
  }
}

// ================================ host =====================================
struct Blk {
  const float *a1_bo, *a1_wk, *a1_wo, *a1_wq, *a1_wv;
  const float *a2_bo, *a2_wk, *a2_wo, *a2_wq, *a2_wv;
  const float *ff_b1, *ff_b2, *ff_w1, *ff_w2;
  const float *ln1_b, *ln1_g, *ln2_b, *ln2_g, *ln3_b, *ln3_g;
};
static Blk loadBlk(void* const* d_in, int i0) {
  Blk k;
  k.a1_bo=(const float*)d_in[i0+0];  k.a1_wk=(const float*)d_in[i0+1];
  k.a1_wo=(const float*)d_in[i0+2];  k.a1_wq=(const float*)d_in[i0+3];
  k.a1_wv=(const float*)d_in[i0+4];
  k.a2_bo=(const float*)d_in[i0+5];  k.a2_wk=(const float*)d_in[i0+6];
  k.a2_wo=(const float*)d_in[i0+7];  k.a2_wq=(const float*)d_in[i0+8];
  k.a2_wv=(const float*)d_in[i0+9];
  k.ff_b1=(const float*)d_in[i0+10]; k.ff_b2=(const float*)d_in[i0+11];
  k.ff_w1=(const float*)d_in[i0+12]; k.ff_w2=(const float*)d_in[i0+13];
  k.ln1_b=(const float*)d_in[i0+14]; k.ln1_g=(const float*)d_in[i0+15];
  k.ln2_b=(const float*)d_in[i0+16]; k.ln2_g=(const float*)d_in[i0+17];
  k.ln3_b=(const float*)d_in[i0+18]; k.ln3_g=(const float*)d_in[i0+19];
  return k;
}

extern "C" void kernel_launch(void* const* d_in, const int* in_sizes, int n_in,
                              void* d_out_v, int out_size, void* d_ws, size_t ws_size,
                              hipStream_t stream) {
  (void)n_in; (void)out_size; (void)ws_size;

  // input index mapping: insertion-order vs fully-sorted pytree flattening
  const bool sortedTop = (in_sizes[0] == B_ * CTX_ * CTXD_);
  int ix, ixr, ictx, iparams, iph, ieot;
  if (!sortedTop) { ix = 0; ixr = 1; ictx = 2; iparams = 3; iph = 49; ieot = 50; }
  else            { ictx = 0; ieot = 1; iparams = 2; iph = 48; ix = 49; ixr = 50; }

  const float* x   = (const float*)d_in[ix];
  const float* xr  = (const float*)d_in[ixr];
  const float* ctx = (const float*)d_in[ictx];
  const int* phIdx  = (const int*)d_in[iph];
  const int* eotIdx = (const int*)d_in[ieot];

  Blk b0  = loadBlk(d_in, iparams + 0);
  const float* gn_b  = (const float*)d_in[iparams + 20];
  const float* gn_g  = (const float*)d_in[iparams + 21];
  Blk img = loadBlk(d_in, iparams + 22);
  const float* pin_b  = (const float*)d_in[iparams + 42];
  const float* pin_w  = (const float*)d_in[iparams + 43];
  const float* pout_b = (const float*)d_in[iparams + 44];
  const float* pout_w = (const float*)d_in[iparams + 45];

  float* out = (float*)d_out_v;
  float* wsf = (float*)d_ws;

  // workspace layout (floats)
  const size_t o_xs  = 0;
  const size_t o_xrs = o_xs  + (size_t)NT_ * D_;
  const size_t o_h   = o_xrs + (size_t)NT_ * D_;
  const size_t o_q   = o_h   + (size_t)NT_ * D_;
  const size_t o_k   = o_q   + (size_t)NT_ * D_;
  const size_t o_v   = o_k   + (size_t)NT_ * D_;
  const size_t o_oh  = o_v   + (size_t)NT_ * D_;
  const size_t o_big = o_oh  + (size_t)NT_ * D_;          // 32*1024*1024 (probs / y / gnt)
  const size_t o_g   = o_big + (size_t)32 * HW_ * HW_;    // geglu out
  const size_t o_pc  = o_g   + (size_t)NT_ * FF2_;        // ctx probs [32,1024,80]
  const size_t o_am  = o_pc  + (size_t)32 * HW_ * DH_;    // attn mean [4,1024,77]
  const size_t o_ph  = o_am  + (size_t)B_ * HW_ * CTX_;
  const size_t o_eot = o_ph  + (size_t)NT_;
  const size_t o_msk = o_eot + (size_t)NT_;
  auto W = [&](size_t o) { return wsf + o; };

  auto launch_gemm = [&](GemmP p, int batch) {
    dim3 g((p.N + 63) / 64, (p.M + 63) / 64, batch);
    gemm_f16wmma<<<g, dim3(128), 0, stream>>>(p);
  };
  auto mk = [](const float* A, int lda, long long sAb, long long sAh,
               const float* Bm, int ldb, long long sBb, long long sBh, int bIsNK,
               float* Dp, int ldd, long long sDb, long long sDh,
               const float* bias, const float* res, int ldres, long long sResb,
               int storeTrans, int M, int N, int K, int heads) {
    GemmP p; p.A = A; p.Bm = Bm; p.bias = bias; p.res = res; p.D = Dp;
    p.M = M; p.N = N; p.K = K; p.lda = lda; p.ldb = ldb; p.ldd = ldd; p.ldres = ldres;
    p.sAb = sAb; p.sAh = sAh; p.sBb = sBb; p.sBh = sBh; p.sDb = sDb; p.sDh = sDh;
    p.sResb = sResb; p.heads = heads; p.bIsNK = bIsNK; p.storeTrans = storeTrans;
    return p;
  };

  const float scale = 0.11180339887498949f; // 80^-0.5

  // attention: q from qin[NT,640]; k/v from kvin[B*mt,kd]; output += into io
  auto attention = [&](const float* qin, const float* kvin, int mt, int kd,
                       const float* wq, const float* wk, const float* wv,
                       const float* wo, const float* bo, float* io,
                       float* probs, int pld, const int* mask, float* meanOut) {
    launch_gemm(mk(qin, D_, 0, 0, wq, D_, 0, 0, 0, W(o_q), D_, 0, 0,
                   nullptr, nullptr, 0, 0, 0, NT_, D_, D_, 1), 1);
    launch_gemm(mk(kvin, kd, 0, 0, wk, D_, 0, 0, 0, W(o_k), D_, 0, 0,
                   nullptr, nullptr, 0, 0, 0, B_ * mt, D_, kd, 1), 1);
    launch_gemm(mk(kvin, kd, 0, 0, wv, D_, 0, 0, 0, W(o_v), D_, 0, 0,
                   nullptr, nullptr, 0, 0, 0, B_ * mt, D_, kd, 1), 1);
    // sim = q @ k^T  (per b,h): A[1024,80], B as NxK from k rows
    launch_gemm(mk(W(o_q), D_, (long long)HW_ * D_, DH_,
                   W(o_k), D_, (long long)mt * D_, DH_, 1,
                   probs, pld, (long long)HEADS_ * HW_ * pld, (long long)HW_ * pld,
                   nullptr, nullptr, 0, 0, 0, HW_, mt, DH_, HEADS_), B_ * HEADS_);
    softmax_rows<<<dim3(B_ * HEADS_ * HW_), dim3(256), 0, stream>>>(
        probs, mt, pld, scale, mask, HEADS_ * HW_, mt);
    if (meanOut) {
      long long tot = (long long)B_ * HW_ * CTX_;
      attn_mean_k<<<dim3((unsigned)((tot + 255) / 256)), dim3(256), 0, stream>>>(probs, meanOut);
    }
    // o = probs @ v  (per b,h): A[1024,mt], B[mt,80]
    launch_gemm(mk(probs, pld, (long long)HEADS_ * HW_ * pld, (long long)HW_ * pld,
                   W(o_v), D_, (long long)mt * D_, DH_, 0,
                   W(o_oh), D_, (long long)HW_ * D_, DH_,
                   nullptr, nullptr, 0, 0, 0, HW_, DH_, mt, HEADS_), B_ * HEADS_);
    // out projection + residual (in place on io)
    launch_gemm(mk(W(o_oh), D_, 0, 0, wo, D_, 0, 0, 0, io, D_, 0, 0,
                   bo, io, D_, 0, 0, NT_, D_, D_, 1), 1);
  };

  auto runBlock = [&](float* io, const Blk& bk, const float* ctxPtr, int mt, int kd,
                      const int* mask, float* meanOut, float* probsCtx, int pldCtx) {
    ln_rows<<<dim3(NT_), dim3(256), 0, stream>>>(io, bk.ln1_g, bk.ln1_b, W(o_h), D_);
    attention(W(o_h), W(o_h), HW_, D_, bk.a1_wq, bk.a1_wk, bk.a1_wv, bk.a1_wo, bk.a1_bo,
              io, W(o_big), HW_, nullptr, nullptr);
    ln_rows<<<dim3(NT_), dim3(256), 0, stream>>>(io, bk.ln2_g, bk.ln2_b, W(o_h), D_);
    attention(W(o_h), ctxPtr, mt, kd, bk.a2_wq, bk.a2_wk, bk.a2_wv, bk.a2_wo, bk.a2_bo,
              io, probsCtx, pldCtx, mask, meanOut);
    ln_rows<<<dim3(NT_), dim3(256), 0, stream>>>(io, bk.ln3_g, bk.ln3_b, W(o_h), D_);
    // FF: y = h @ w1 + b1  ([4096,5120] into the big region)
    launch_gemm(mk(W(o_h), D_, 0, 0, bk.ff_w1, FF1_, 0, 0, 0, W(o_big), FF1_, 0, 0,
                   bk.ff_b1, nullptr, 0, 0, 0, NT_, FF1_, D_, 1), 1);
    long long tot = (long long)NT_ * FF2_;
    geglu_k<<<dim3((unsigned)((tot + 255) / 256)), dim3(256), 0, stream>>>(W(o_big), W(o_g), tot);
    launch_gemm(mk(W(o_g), FF2_, 0, 0, bk.ff_w2, D_, 0, 0, 0, io, D_, 0, 0,
                   bk.ff_b2, io, D_, 0, 0, NT_, D_, FF2_, 1), 1);
  };

  // ---- proj_in: GN+transpose -> 1x1 conv GEMM ----
  gn_transpose<<<dim3(B_ * 32), dim3(256), 0, stream>>>(x, gn_g, gn_b, W(o_big));
  launch_gemm(mk(W(o_big), C_, 0, 0, pin_w, D_, 0, 0, 0, W(o_xs), D_, 0, 0,
                 pin_b, nullptr, 0, 0, 0, NT_, D_, C_, 1), 1);
  gn_transpose<<<dim3(B_ * 32), dim3(256), 0, stream>>>(xr, gn_g, gn_b, W(o_big));
  launch_gemm(mk(W(o_big), C_, 0, 0, pin_w, D_, 0, 0, 0, W(o_xrs), D_, 0, 0,
                 pin_b, nullptr, 0, 0, 0, NT_, D_, C_, 1), 1);

  // ---- block0 on xs and xrs (xrs pass returns head-mean attention) ----
  runBlock(W(o_xs),  b0, ctx, CTX_, CTXD_, nullptr, nullptr,   W(o_pc), DH_);
  runBlock(W(o_xrs), b0, ctx, CTX_, CTXD_, nullptr, W(o_am),   W(o_pc), DH_);

  // ---- loss + otsu mask ----
  gather_ph_k<<<dim3(NT_ / 256), dim3(256), 0, stream>>>(W(o_am), phIdx, eotIdx, W(o_ph), W(o_eot));
  loss_k<<<dim3(1), dim3(256), 0, stream>>>(W(o_ph), W(o_eot),
                                            out + (size_t)2 * B_ * C_ * HW_);
  otsu_mask_k<<<dim3(B_), dim3(256), 0, stream>>>(W(o_ph), (int*)W(o_msk));

  // ---- imgblock on xs with ctx = xrs, masked attention ----
  runBlock(W(o_xs), img, W(o_xrs), HW_, D_, (const int*)W(o_msk), nullptr, W(o_big), HW_);

  // ---- proj_out (transposed store + bias + input residual) ----
  launch_gemm(mk(W(o_xs), D_, (long long)HW_ * D_, 0,
                 pout_w, C_, 0, 0, 0,
                 out, HW_, (long long)C_ * HW_, 0,
                 pout_b, x, HW_, (long long)C_ * HW_, 1,
                 HW_, C_, D_, 1), B_);
  launch_gemm(mk(W(o_xrs), D_, (long long)HW_ * D_, 0,
                 pout_w, C_, 0, 0, 0,
                 out + (size_t)B_ * C_ * HW_, HW_, (long long)C_ * HW_, 0,
                 pout_b, xr, HW_, (long long)C_ * HW_, 1,
                 HW_, C_, D_, 1), B_);
}
